// SpatialConvolution_71554155151994
// MI455X (gfx1250) — compile-verified, matrix operand order checked by reading, not run on hardware
//
#include <hip/hip_runtime.h>
#include <hip/hip_bf16.h>

// ---------------------------------------------------------------------------
// SpatialConvolution for MI455X (gfx1250, wave32, WMMA).
//  K0: zero f32 accumulator in workspace
//  K1: edge MLP  -> sw[E,16] f16 (= (silu(ea@W1+b1)@W2+b2) * edge_sh, padded)
//  K2: msg GEMM  X[e, m*64+f] = h_src[e,f]*sw[e,m];  msg = X @ Wtp'(576x64)
//      via v_wmma_f32_16x16x32_f16, scatter-add to agg via atomic fadd
//  K3: FiLM + degree norm -> d_out
// ---------------------------------------------------------------------------

typedef __attribute__((ext_vector_type(16))) _Float16 v16h;
typedef __attribute__((ext_vector_type(8)))  _Float16 v8h;
typedef __attribute__((ext_vector_type(8)))  float    v8f;

#define WMMA_F32_F16(a, b, c) \
    __builtin_amdgcn_wmma_f32_16x16x32_f16(false, (a), false, (b), (short)0, (c), false, false)

// fast silu: v_exp_f32 (TRANS, co-executes with XDL) + v_rcp_f32 + mul,
// instead of the IEEE divide expansion (~12 VALU ops).
__device__ __forceinline__ float silu_f(float x) {
    return x * __builtin_amdgcn_rcpf(1.0f + __expf(-x));
}

// Build a v16h fragment from two aligned 8-half groups in LDS.
// ISA layout: element j <-> K = (j/8)*16 + (lane/16)*8 + (j%8)  (same map for A and B).
__device__ __forceinline__ v16h frag_from_lds(const _Float16* row, int kbase, int ksel8) {
    const v8h lo = *(const v8h*)(row + kbase + ksel8);
    const v8h hi = *(const v8h*)(row + kbase + 16 + ksel8);
    v16h r;
#pragma unroll
    for (int i = 0; i < 8; ++i) { r[i] = lo[i]; r[8 + i] = hi[i]; }
    return r;
}

// ---------------------------------------------------------------------------
// K0: zero fill
// ---------------------------------------------------------------------------
__global__ __launch_bounds__(256) void k0_zero(float* __restrict__ p, int n) {
    int i = blockIdx.x * blockDim.x + threadIdx.x;
    int stride = gridDim.x * blockDim.x;
    for (; i < n; i += stride) p[i] = 0.0f;
}

// ---------------------------------------------------------------------------
// K1: edge-weight MLP, WMMA.  128 threads = 4 waves, 16 edges per wave.
// ---------------------------------------------------------------------------
__global__ __launch_bounds__(128) void k1_edge_mlp(
    const float* __restrict__ edge_attr,   // [E,32]
    const float* __restrict__ edge_sh,     // [E,9]
    const float* __restrict__ W1,          // [32,64]
    const float* __restrict__ b1,          // [64]
    const float* __restrict__ W2,          // [64,9]
    const float* __restrict__ b2,          // [9]
    _Float16* __restrict__ swbuf,          // [E,16] out
    int E)
{
    __shared__ __align__(16) _Float16 sW1t[64 * 32];      // [h][k]  (B^T for GEMM1)
    __shared__ __align__(16) _Float16 sW2t[16 * 64];      // [g][k]  (B^T for GEMM2, padded 9->16)
    __shared__ __align__(16) _Float16 sEA[4][16][32];     // per-wave A tile, GEMM1
    __shared__ __align__(16) _Float16 sHID[4][16][64];    // per-wave A tile, GEMM2

    const int tid  = threadIdx.x;
    const int wid  = tid >> 5;
    const int lane = tid & 31;
    const int col  = lane & 15;
    const int ksel = (lane >> 4) & 1;       // 0: lanes 0-15, 1: lanes 16-31
    const int ksel8 = ksel * 8;

    // cooperative weight staging (transposed, f32 -> f16)
    for (int t = tid; t < 64 * 32; t += 128) {
        int h = t >> 5, k = t & 31;
        sW1t[t] = (_Float16)W1[k * 64 + h];
    }
    for (int t = tid; t < 16 * 64; t += 128) {
        int g = t >> 6, k = t & 63;
        sW2t[t] = (g < 9) ? (_Float16)W2[k * 9 + g] : (_Float16)0.0f;
    }

    // stage edge_attr tile (16 rows x 32 f32 -> f16), 2 lanes per row
    const int e0  = blockIdx.x * 64 + wid * 16;
    const int row = lane >> 1;
    const int hlf = lane & 1;
    {
        _Float16 tmp[16];
        const int e = e0 + row;
        if (e < E) {
            const float4* p = (const float4*)(edge_attr + (size_t)e * 32 + hlf * 16);
#pragma unroll
            for (int i = 0; i < 4; ++i) {
                float4 v = p[i];
                tmp[i * 4 + 0] = (_Float16)v.x; tmp[i * 4 + 1] = (_Float16)v.y;
                tmp[i * 4 + 2] = (_Float16)v.z; tmp[i * 4 + 3] = (_Float16)v.w;
            }
        } else {
#pragma unroll
            for (int i = 0; i < 16; ++i) tmp[i] = (_Float16)0.0f;
        }
        *(v8h*)(&sEA[wid][row][hlf * 16])     = *(const v8h*)(tmp);
        *(v8h*)(&sEA[wid][row][hlf * 16 + 8]) = *(const v8h*)(tmp + 8);
    }
    __syncthreads();

    // GEMM1: hidden(16x64) = EA(16x32) @ W1(32x64) + b1   (1 k-step, 4 n-tiles)
    v8f acc[4];
#pragma unroll
    for (int n = 0; n < 4; ++n) {
        float bb = b1[n * 16 + col];
#pragma unroll
        for (int r = 0; r < 8; ++r) acc[n][r] = bb;
    }
    {
        const v16h a = frag_from_lds(&sEA[wid][col][0], 0, ksel8);
#pragma unroll
        for (int n = 0; n < 4; ++n) {
            const v16h b = frag_from_lds(&sW1t[(n * 16 + col) * 32], 0, ksel8);
            acc[n] = WMMA_F32_F16(a, b, acc[n]);
        }
    }
    // silu -> f16 hidden tile
#pragma unroll
    for (int n = 0; n < 4; ++n) {
#pragma unroll
        for (int r = 0; r < 8; ++r) {
            sHID[wid][r + ksel8][n * 16 + col] = (_Float16)silu_f(acc[n][r]);
        }
    }
    __syncthreads();

    // GEMM2: wvec(16x16) = HID(16x64) @ W2t'(64x16) + b2   (2 k-steps)
    v8f acc2;
    {
        float bb = (col < 9) ? b2[col] : 0.0f;
#pragma unroll
        for (int r = 0; r < 8; ++r) acc2[r] = bb;
    }
#pragma unroll
    for (int ks = 0; ks < 2; ++ks) {
        const v16h a = frag_from_lds(&sHID[wid][col][0], ks * 32, ksel8);
        const v16h b = frag_from_lds(&sW2t[col * 64], ks * 32, ksel8);
        acc2 = WMMA_F32_F16(a, b, acc2);
    }
    // multiply by edge_sh, store padded f16 sw
#pragma unroll
    for (int r = 0; r < 8; ++r) {
        const int e = e0 + r + ksel8;
        if (e < E) {
            float shv = (col < 9) ? edge_sh[(size_t)e * 9 + col] : 0.0f;
            swbuf[(size_t)e * 16 + col] = (_Float16)(acc2[r] * shv);
        }
    }
}

// ---------------------------------------------------------------------------
// K2: weighted tensor product + scatter.  128 threads = 4 waves, 16 edges/wave.
//     msg(16x64) = X(16x576) @ Bt(576x64);  X[e, m*64+f] = h[e,f] * sw[e,m]
//     Bt chunked through LDS in 3 x (64 x 192) f16 tiles (24 KB each).
// ---------------------------------------------------------------------------
__global__ __launch_bounds__(128) void k2_tensor_product(
    const float* __restrict__ features,    // [N,64]
    const int*   __restrict__ src,         // [E]
    const int*   __restrict__ dst,         // [E]
    const _Float16* __restrict__ swbuf,    // [E,16]
    const float* __restrict__ W_tp,        // [64,9,64]
    float* __restrict__ agg,               // [N,64] atomic accumulator
    int E)
{
    __shared__ __align__(16) _Float16 sB[64 * 192];     // Bt chunk: [g][kk]
    __shared__ __align__(16) _Float16 sH[4][16][64];    // gathered h_src tile, f16
    __shared__ __align__(16) _Float16 sSW[4][16][16];   // sw tile

    const int tid  = threadIdx.x;
    const int wid  = tid >> 5;
    const int lane = tid & 31;
    const int col  = lane & 15;
    const int ksel = (lane >> 4) & 1;
    const int ksel8 = ksel * 8;

    const int e0  = blockIdx.x * 64 + wid * 16;
    const int row = lane >> 1;
    const int hlf = lane & 1;

    // ---- stage h_src (gather, f32->f16) and sw tiles, 2 lanes per edge row
    {
        _Float16 tmp[32];
        const int e = e0 + row;
        if (e < E) {
            const int sidx = src[e];
            const float4* p = (const float4*)(features + (size_t)sidx * 64 + hlf * 32);
#pragma unroll
            for (int i = 0; i < 8; ++i) {
                float4 v = p[i];
                tmp[i * 4 + 0] = (_Float16)v.x; tmp[i * 4 + 1] = (_Float16)v.y;
                tmp[i * 4 + 2] = (_Float16)v.z; tmp[i * 4 + 3] = (_Float16)v.w;
            }
        } else {
#pragma unroll
            for (int i = 0; i < 32; ++i) tmp[i] = (_Float16)0.0f;
        }
#pragma unroll
        for (int i = 0; i < 4; ++i)
            *(v8h*)(&sH[wid][row][hlf * 32 + i * 8]) = *(const v8h*)(tmp + i * 8);

        v8h swv;
        if (e < E) {
            swv = *(const v8h*)(swbuf + (size_t)e * 16 + hlf * 8);
        } else {
#pragma unroll
            for (int i = 0; i < 8; ++i) swv[i] = (_Float16)0.0f;
        }
        *(v8h*)(&sSW[wid][row][hlf * 8]) = swv;
    }

    v8f acc[4];
#pragma unroll
    for (int n = 0; n < 4; ++n)
#pragma unroll
        for (int r = 0; r < 8; ++r) acc[n][r] = 0.0f;

    // ---- 3 chunks of K: each chunk covers k in [c*192, c*192+192) = 3 m-values
    for (int c = 0; c < 3; ++c) {
        __syncthreads();   // previous chunk consumed / tiles staged
        // stage Bt chunk: sB[g*192 + j] = W_tp[f,m,g], k = c*192+j, f=k&63, m=k>>6
        // vectorized along g (contiguous in W_tp): float4 load + 4 scattered b16 stores
        for (int t = tid; t < (64 * 192) / 4; t += 128) {
            const int g0 = (t & 15) * 4;          // 0,4,...,60
            const int j  = t >> 4;                // 0..191
            const int k  = c * 192 + j;
            const float4 v = *(const float4*)(W_tp + (size_t)(k & 63) * 576
                                                   + (size_t)(k >> 6) * 64 + g0);
            sB[(g0 + 0) * 192 + j] = (_Float16)v.x;
            sB[(g0 + 1) * 192 + j] = (_Float16)v.y;
            sB[(g0 + 2) * 192 + j] = (_Float16)v.z;
            sB[(g0 + 3) * 192 + j] = (_Float16)v.w;
        }
        __syncthreads();

#pragma unroll
        for (int ks = 0; ks < 6; ++ks) {
            // A fragment: X[row, k] = h[row, k&63] * sw[row, k>>6]
            const int kAlo = c * 192 + ks * 32 + ksel8;       // absolute k, elements 0..7
            const int kAhi = kAlo + 16;                        // elements 8..15
            const _Float16* hrow  = &sH[wid][col][0];
            const _Float16* swrow = &sSW[wid][col][0];
            const v8h hlo = *(const v8h*)(hrow + (kAlo & 63));
            const v8h hhi = *(const v8h*)(hrow + (kAhi & 63));
            const _Float16 slo = swrow[kAlo >> 6];
            const _Float16 shi = swrow[kAhi >> 6];
            v16h a;
#pragma unroll
            for (int i = 0; i < 8; ++i) { a[i] = hlo[i] * slo; a[8 + i] = hhi[i] * shi; }

#pragma unroll
            for (int n = 0; n < 4; ++n) {
                const v16h b = frag_from_lds(&sB[(n * 16 + col) * 192], ks * 32, ksel8);
                acc[n] = WMMA_F32_F16(a, b, acc[n]);
            }
        }
    }

    // ---- scatter-add rows to destination nodes
#pragma unroll
    for (int n = 0; n < 4; ++n) {
#pragma unroll
        for (int r = 0; r < 8; ++r) {
            const int e = e0 + r + ksel8;
            if (e < E) {
                const int d = dst[e];
                unsafeAtomicAdd(&agg[(size_t)d * 64 + n * 16 + col], acc[n][r]);
            }
        }
    }
}

// ---------------------------------------------------------------------------
// K3: FiLM + degree normalization.  256 threads = 4 nodes x 64 features.
// ---------------------------------------------------------------------------
__global__ __launch_bounds__(256) void k3_film(
    const float* __restrict__ agg,     // [N,64]
    const float* __restrict__ c_noise, // [N]
    const float* __restrict__ Wn1,     // [1,64]
    const float* __restrict__ bn1,     // [64]
    const float* __restrict__ Wn2,     // [64,128]
    const float* __restrict__ bn2,     // [128]
    float* __restrict__ out,           // [N,64]
    int N, float scale)
{
    __shared__ float sHid[4][64];
    const int nl = threadIdx.x >> 6;
    const int g  = threadIdx.x & 63;
    const int n  = blockIdx.x * 4 + nl;

    float c = (n < N) ? c_noise[n] : 0.0f;
    sHid[nl][g] = silu_f(c * Wn1[g] + bn1[g]);
    __syncthreads();

    if (n >= N) return;
    float gamma = 0.0f, beta = 0.0f;
#pragma unroll 8
    for (int h = 0; h < 64; ++h) {
        const float hv = sHid[nl][h];
        gamma = fmaf(hv, Wn2[h * 128 + g], gamma);
        beta  = fmaf(hv, Wn2[h * 128 + 64 + g], beta);
    }
    gamma += bn2[g];
    beta  += bn2[64 + g];
    const float a = agg[(size_t)n * 64 + g] * scale;
    out[(size_t)n * 64 + g] = a * (1.0f + gamma) + beta;
}

// ---------------------------------------------------------------------------
// host launcher
// ---------------------------------------------------------------------------
extern "C" void kernel_launch(void* const* d_in, const int* in_sizes, int n_in,
                              void* d_out, int out_size, void* d_ws, size_t ws_size,
                              hipStream_t stream) {
    const float* features  = (const float*)d_in[0];
    const int*   edge_idx  = (const int*)  d_in[1];
    const float* edge_attr = (const float*)d_in[2];
    const float* edge_sh   = (const float*)d_in[3];
    const float* c_noise   = (const float*)d_in[4];
    const float* W1        = (const float*)d_in[5];
    const float* b1        = (const float*)d_in[6];
    const float* W2        = (const float*)d_in[7];
    const float* b2        = (const float*)d_in[8];
    const float* W_tp      = (const float*)d_in[9];
    const float* Wn1       = (const float*)d_in[10];
    const float* bn1       = (const float*)d_in[11];
    const float* Wn2       = (const float*)d_in[12];
    const float* bn2       = (const float*)d_in[13];

    const int N = in_sizes[0] / 64;     // features [N,64]
    const int E = in_sizes[2] / 32;     // edge_attr [E,32]
    const int* src = edge_idx;
    const int* dst = edge_idx + E;

    // workspace layout: [ sw f16 : E*16 ][ agg f32 : N*64 ]
    _Float16* swbuf = (_Float16*)d_ws;
    size_t sw_bytes = (((size_t)E * 16 * sizeof(_Float16)) + 255) & ~(size_t)255;
    float* agg = (float*)((char*)d_ws + sw_bytes);

    const float scale = 1.0f / sqrtf((float)E / (float)N);   // 1/sqrt(avg_deg)

    const int edge_blocks = (E + 63) / 64;
    const int node_blocks = (N + 3) / 4;

    k0_zero<<<(N * 64 + 255) / 256, 256, 0, stream>>>(agg, N * 64);
    k1_edge_mlp<<<edge_blocks, 128, 0, stream>>>(edge_attr, edge_sh, W1, b1, W2, b2,
                                                 swbuf, E);
    k2_tensor_product<<<edge_blocks, 128, 0, stream>>>(features, src, dst, swbuf,
                                                       W_tp, agg, E);
    k3_film<<<node_blocks, 256, 0, stream>>>(agg, c_noise, Wn1, bn1, Wn2, bn2,
                                             (float*)d_out, N, scale);
}
